// SiameseNetwork_55817394979539
// MI455X (gfx1250) — matrix operand order repaired and checked by reading, not compiled, not run on hardware
//
#include <hip/hip_runtime.h>
#include <hip/hip_bf16.h>

// ---------------------------------------------------------------------------
// CDNA5 (gfx1250) WMMA implicit-GEMM Siamese CNN — bf16 activations/weights,
// fp32 accumulate via V_WMMA_F32_16X16X32_BF16.
//   Block tile: 64 (Cout) x 128 (pixels), K staged 64-deep.
//   Double-buffered LDS pipeline: stage k+1 loads overlap stage k WMMAs,
//   one barrier per stage. Weight staging uses GLOBAL_LOAD_ASYNC_TO_LDS_B128
//   (ASYNCcnt) when the toolchain exposes the builtin.
// ---------------------------------------------------------------------------

typedef __attribute__((ext_vector_type(16))) __bf16 v16bf;
typedef __attribute__((ext_vector_type(8)))  __bf16 v8bf;
typedef __attribute__((ext_vector_type(8)))  float  v8f;
typedef __attribute__((__vector_size__(16))) int    i32x4;

#define GLOBAL_AS __attribute__((address_space(1)))
#define LDS_AS    __attribute__((address_space(3)))

#if __has_builtin(__builtin_amdgcn_global_load_async_to_lds_b128)
#define USE_ASYNC_LDS 1
#else
#define USE_ASYNC_LDS 0
#endif

__device__ __forceinline__ void copy16_g2l(const __bf16* g, __bf16* l) {
#if USE_ASYNC_LDS
    __builtin_amdgcn_global_load_async_to_lds_b128(
        (GLOBAL_AS i32x4*)g, (LDS_AS i32x4*)l, 0, 0);
#else
    *(uint4*)l = *(const uint4*)g;
#endif
}

__device__ __forceinline__ void wait_async_lds() {
#if USE_ASYNC_LDS
#if __has_builtin(__builtin_amdgcn_s_wait_asynccnt)
    __builtin_amdgcn_s_wait_asynccnt(0);
#else
    asm volatile("s_wait_asynccnt 0x0" ::: "memory");
#endif
#endif
}

__device__ __forceinline__ __bf16 bf_zero() {
    return __builtin_bit_cast(__bf16, (unsigned short)0);
}

__device__ __forceinline__ __bf16 f2bf(float f) {
    unsigned u = __builtin_bit_cast(unsigned, f);
    u = (u + 0x7FFFu + ((u >> 16) & 1u)) >> 16;   // round-to-nearest-even
    return __builtin_bit_cast(__bf16, (unsigned short)u);
}

// A fragment (16x32 of a [row][64] LDS tile, k-substep ks in {0,1}):
// lanes 0-15 : M=lane,    K = 0..15  (+32*ks)
// lanes 16-31: M=lane-16, K = 16..31 (+32*ks)
__device__ __forceinline__ v16bf frag_A(const __bf16* base, int lane, int ks) {
    const __bf16* p = base + (lane & 15) * 64 + ks * 32 + (lane & 16);
    v8bf lo = *(const v8bf*)p;          // 16B aligned
    v8bf hi = *(const v8bf*)(p + 8);    // 16B aligned
    return __builtin_shufflevector(lo, hi, 0, 1, 2, 3, 4, 5, 6, 7,
                                           8, 9, 10, 11, 12, 13, 14, 15);
}

// B fragment (32x16 of a [col][64] LDS tile):
// lanes 0-15 : N=lane,    K = 0..15  (+32*ks)
// lanes 16-31: N=lane-16, K = 16..31 (+32*ks)
__device__ __forceinline__ v16bf frag_B(const __bf16* base, int lane, int ks) {
    const __bf16* p = base + (lane & 15) * 64 + ks * 32 + (lane & 16);
    return *(const v16bf*)p;            // 32B aligned
}

// ---------------------------------------------------------------------------
// Implicit-GEMM conv, stride 2, SAME (pad_lo = 0 for even inputs).
//   y[N,Cout,Ho,Wo] = relu( W[Cout,K] x im2col(x) ),  K = Cin*9 (Kp padded)
// ---------------------------------------------------------------------------
__global__ __launch_bounds__(256)
void conv_wmma(const __bf16* __restrict__ x, const __bf16* __restrict__ wb,
               __bf16* __restrict__ y,
               int Cin, int Hin, int Win,
               int Cout, int Hout, int Wout, int K, int Kp) {
    __shared__ __attribute__((aligned(32))) __bf16 lA[2][64 * 64];   // [co][k]
    __shared__ __attribute__((aligned(32))) __bf16 lB[2][128 * 64];  // [pix][k]
    __shared__ int colBase[128], colImg[128], colOh[128], colOw[128];

    const int tid  = threadIdx.x;
    const int lane = tid & 31;
    const int wave = tid >> 5;
    const int mi   = wave & 3;              // M subtile (0..3)
    const int nbB  = (wave >> 2) * 4;       // first of 4 N subtiles

    const int nBase  = blockIdx.x * 128;
    const int mBase  = blockIdx.y * 64;
    const int HW     = Hout * Wout;
    const int HinWin = Hin * Win;

    // ---- per-column im2col descriptors: one decode per pixel per block ----
    if (tid < 128) {
        int p   = nBase + tid;
        int img = p / HW;
        int rem = p - img * HW;
        int oh  = rem / Wout;
        int ow  = rem - oh * Wout;
        colImg[tid]  = img;
        colOh[tid]   = oh;
        colOw[tid]   = ow;
        colBase[tid] = (img * Cin * Hin + oh * 2) * Win + ow * 2;
    }
    __syncthreads();

    const int kloc  = tid & 63;      // this thread's fixed k lane in the tile
    const int nrow0 = tid >> 6;      // first pixel row this thread fills

    // ---- stage loaders ----
    auto stageA = [&](int kt, __bf16* dstA) {
        const int k0 = kt * 64;
#pragma unroll
        for (int i = 0; i < 2; ++i) {
            int e8 = tid + i * 256;                 // 8-element group id
            int m  = e8 >> 3, kg = (e8 & 7) * 8;
            copy16_g2l(wb + (mBase + m) * Kp + k0 + kg, dstA + m * 64 + kg);
        }
    };
    auto stageB = [&](int kt, __bf16* dstB) {
        const int  kk     = kt * 64 + kloc;
        const bool kvalid = kk < K;
        int r = 0, s = 0, coff = 0;
        if (kvalid) {
            int c  = kk / 9;                        // constant divisors ->
            int rs = kk - c * 9;                    // mul/shift, no div expand
            r      = rs / 3;
            s      = rs - r * 3;
            coff   = c * HinWin + r * Win + s;
        }
#pragma unroll
        for (int i = 0; i < 32; ++i) {
            int n = nrow0 + i * 4;
            __bf16 v = bf_zero();
            if (kvalid) {
                int iy = colOh[n] * 2 + r;
                int ix = colOw[n] * 2 + s;
                if (iy < Hin && ix < Win)
                    v = x[colBase[n] + coff];
            }
            dstB[n * 64 + kloc] = v;
        }
    };

    v8f acc[4] = {};
    const int KT = Kp >> 6;

    // ---- prologue: fill stage 0 ----
    stageA(0, lA[0]);
    stageB(0, lB[0]);
    wait_async_lds();
    __syncthreads();

    // ---- double-buffered main loop: one barrier per stage ----
    for (int kt = 0; kt < KT; ++kt) {
        const int     cur  = kt & 1;
        const __bf16* curA = lA[cur];
        const __bf16* curB = lB[cur];
        if (kt + 1 < KT) {                    // issue next stage's loads
            stageA(kt + 1, lA[cur ^ 1]);
            stageB(kt + 1, lB[cur ^ 1]);
        }
        // 8 WMMAs per wave per stage, overlapped with the loads above
#pragma unroll
        for (int ks = 0; ks < 2; ++ks) {
            v16bf a = frag_A(curA + mi * 16 * 64, lane, ks);
#pragma unroll
            for (int t = 0; t < 4; ++t) {
                v16bf b = frag_B(curB + (nbB + t) * 16 * 64, lane, ks);
                acc[t] = __builtin_amdgcn_wmma_f32_16x16x32_bf16(
                    false, a, false, b, (short)0, acc[t], false, false);
            }
        }
        if (kt + 1 < KT) {
            wait_async_lds();
            __syncthreads();
        }
    }

    // ---- epilogue: ReLU + bf16 scatter back to NCHW ----
    const int nl = lane & 15;
    const int mb = (lane & 16) >> 1;    // 0 or 8
#pragma unroll
    for (int t = 0; t < 4; ++t) {
        int n   = (nbB + t) * 16 + nl;
        int img = colImg[n], oh = colOh[n], ow = colOw[n];
        __bf16* yp = y + (img * Cout * Hout + oh) * Wout + ow;
#pragma unroll
        for (int r = 0; r < 8; ++r) {
            int   co = mBase + mi * 16 + mb + r;
            float v  = acc[t][r];
            v = v > 0.f ? v : 0.f;
            yp[co * HW] = f2bf(v);
        }
    }
}

// ---------------------------------------------------------------------------
// Head GEMM (single block): out[64,128] = pooled[64,512] x wl[512,128] + bl
// ---------------------------------------------------------------------------
__global__ __launch_bounds__(256)
void head_wmma(const float* __restrict__ pooled, const __bf16* __restrict__ wlb,
               const float* __restrict__ bl, float* __restrict__ out) {
    __shared__ __attribute__((aligned(32))) __bf16 lA[2][64 * 64];   // [img][k]
    __shared__ __attribute__((aligned(32))) __bf16 lB[2][128 * 64];  // [feat][k]

    const int tid  = threadIdx.x;
    const int lane = tid & 31;
    const int wave = tid >> 5;
    const int mi   = wave & 3;
    const int nbB  = (wave >> 2) * 4;
    const int K    = 512;

    const int kloc  = tid & 63;
    const int nrow0 = tid >> 6;

    auto stage = [&](int kt, __bf16* dstA, __bf16* dstB) {
        const int k0 = kt * 64;
#pragma unroll
        for (int i = 0; i < 16; ++i) {           // A: 64x64, fp32 -> bf16
            int m = nrow0 + i * 4;
            dstA[m * 64 + kloc] = f2bf(pooled[m * K + k0 + kloc]);
        }
#pragma unroll
        for (int i = 0; i < 32; ++i) {           // B: 128x64, bf16 gather
            int n = nrow0 + i * 4;
            dstB[n * 64 + kloc] = wlb[(k0 + kloc) * 128 + n];
        }
    };

    v8f acc[4] = {};
    const int KT = K / 64;

    stage(0, lA[0], lB[0]);
    __syncthreads();

    for (int kt = 0; kt < KT; ++kt) {
        const int     cur  = kt & 1;
        const __bf16* curA = lA[cur];
        const __bf16* curB = lB[cur];
        if (kt + 1 < KT) stage(kt + 1, lA[cur ^ 1], lB[cur ^ 1]);
#pragma unroll
        for (int ks = 0; ks < 2; ++ks) {
            v16bf a = frag_A(curA + mi * 16 * 64, lane, ks);
#pragma unroll
            for (int t = 0; t < 4; ++t) {
                v16bf b = frag_B(curB + (nbB + t) * 16 * 64, lane, ks);
                acc[t] = __builtin_amdgcn_wmma_f32_16x16x32_bf16(
                    false, a, false, b, (short)0, acc[t], false, false);
            }
        }
        if (kt + 1 < KT) __syncthreads();
    }

    const int nl = lane & 15;
    const int mb = (lane & 16) >> 1;
#pragma unroll
    for (int t = 0; t < 4; ++t) {
        int   col  = (nbB + t) * 16 + nl;
        float bias = bl[col];
#pragma unroll
        for (int r = 0; r < 8; ++r) {
            int row = mi * 16 + mb + r;          // image index
            out[row * 128 + col] = acc[t][r] + bias;
        }
    }
}

// ---------------------------------------------------------------------------
// Global average pool: bf16 [N,512,14,14] -> fp32 [N,512]  (HW = 196 fixed)
// ---------------------------------------------------------------------------
__global__ __launch_bounds__(256)
void avgpool_k(const __bf16* __restrict__ x, float* __restrict__ y, int total) {
    int i = blockIdx.x * blockDim.x + threadIdx.x;
    if (i >= total) return;
    const __bf16* p = x + (long long)i * 196;
    float s = 0.f;
#pragma unroll 4
    for (int j = 0; j < 196; ++j) s += (float)p[j];
    y[i] = s * (1.f / 196.f);
}

// ---------------------------------------------------------------------------
// fp32 -> bf16 convert (flat layout preserved)
// ---------------------------------------------------------------------------
__global__ __launch_bounds__(256)
void cvt_k(const float* __restrict__ src, __bf16* __restrict__ dst, int total) {
    int i = blockIdx.x * blockDim.x + threadIdx.x;
    if (i < total) dst[i] = f2bf(src[i]);
}

// Weight convert with K padding (2D grid: y = row): no integer division.
__global__ __launch_bounds__(256)
void cvt_w_k(const float* __restrict__ src, __bf16* __restrict__ dst,
             int K, int Kp) {
    int k   = blockIdx.x * blockDim.x + threadIdx.x;
    int row = blockIdx.y;
    if (k >= Kp) return;
    dst[row * Kp + k] = f2bf((k < K) ? src[row * K + k] : 0.f);
}

// input1_[img,i] = flag(img) != 0 ? flag(img) : input2[img,i]  (bf16 output)
// 2D grid: y = image -> no integer division.
__global__ __launch_bounds__(256)
void select_bf_k(const float* __restrict__ in2, const long long* __restrict__ flag,
                 __bf16* __restrict__ out, int perImg) {
    int i   = blockIdx.x * blockDim.x + threadIdx.x;
    int img = blockIdx.y;
    if (i >= perImg) return;
    float f = (float)flag[img];
    long long idx = (long long)img * perImg + i;
    out[idx] = f2bf((f != 0.f) ? f : in2[idx]);
}

// ---------------------------------------------------------------------------
extern "C" void kernel_launch(void* const* d_in, const int* in_sizes, int n_in,
                              void* d_out, int out_size, void* d_ws, size_t ws_size,
                              hipStream_t stream) {
    (void)in_sizes; (void)n_in; (void)out_size; (void)ws_size;

    const float*     in1 = (const float*)d_in[0];
    const float*     in2 = (const float*)d_in[1];
    const long long* in3 = (const long long*)d_in[2];
    const float*     w1  = (const float*)d_in[3];
    const float*     w2  = (const float*)d_in[4];
    const float*     w3  = (const float*)d_in[5];
    const float*     w4  = (const float*)d_in[6];
    const float*     wl  = (const float*)d_in[7];
    const float*     bl  = (const float*)d_in[8];
    float* out = (float*)d_out;

    // -------- workspace layout (bf16 elements; offsets stay 16B aligned) ----
    __bf16* wsb  = (__bf16*)d_ws;
    __bf16* xbf  = wsb;                         //  9,633,792  (input, bf16)
    __bf16* actA = xbf  + 9633792;              // 51,380,224
    __bf16* actB = actA + 51380224;             // 25,690,112
    __bf16* wb1  = actB + 25690112;             //  64*64      (K=27 -> Kp=64)
    __bf16* wb2  = wb1  + 4096;                 // 128*576
    __bf16* wb3  = wb2  + 73728;                // 256*1152
    __bf16* wb4  = wb3  + 294912;               // 512*2304
    __bf16* wlb  = wb4  + 1179648;              // 512*128
    float*  pooled = (float*)(wlb + 65536);     // 64*512 fp32

    // -------- one-time weight conversion (deterministic every call) --------
    cvt_w_k<<<dim3(1,  64),  256, 0, stream>>>(w1, wb1, 27,   64);
    cvt_w_k<<<dim3(3,  128), 256, 0, stream>>>(w2, wb2, 576,  576);
    cvt_w_k<<<dim3(5,  256), 256, 0, stream>>>(w3, wb3, 1152, 1152);
    cvt_w_k<<<dim3(9,  512), 256, 0, stream>>>(w4, wb4, 2304, 2304);
    cvt_k  <<<(65536 + 255) / 256, 256, 0, stream>>>(wl, wlb, 65536);

    auto run_branch = [&](float* outPtr) {
        conv_wmma<<<dim3(802816 / 128, 1), 256, 0, stream>>>(
            xbf,  wb1, actA, 3,   224, 224, 64,  112, 112, 27,   64);
        conv_wmma<<<dim3(200704 / 128, 2), 256, 0, stream>>>(
            actA, wb2, actB, 64,  112, 112, 128, 56,  56,  576,  576);
        conv_wmma<<<dim3(50176 / 128, 4), 256, 0, stream>>>(
            actB, wb3, actA, 128, 56,  56,  256, 28,  28,  1152, 1152);
        conv_wmma<<<dim3(12544 / 128, 8), 256, 0, stream>>>(
            actA, wb4, actB, 256, 28,  28,  512, 14,  14,  2304, 2304);
        avgpool_k<<<(32768 + 255) / 256, 256, 0, stream>>>(actB, pooled, 32768);
        head_wmma<<<1, 256, 0, stream>>>(pooled, wlb, bl, outPtr);
    };

    const int perImg = 3 * 224 * 224;           // 150,528 = 588 * 256
    const int total  = 64 * perImg;

    // branch 1: input1 -> out[0 .. 8191]
    cvt_k<<<(total + 255) / 256, 256, 0, stream>>>(in1, xbf, total);
    run_branch(out);

    // branch 2: where(flag, flag, input2) -> out[8192 .. 16383]
    select_bf_k<<<dim3(588, 64), 256, 0, stream>>>(in2, in3, xbf, perImg);
    run_branch(out + 64 * 128);
}
